// E8RHTFusedExperts_5763846111361
// MI455X (gfx1250) — matrix operand order repaired and checked by reading, not compiled, not run on hardware
//
#include <hip/hip_runtime.h>

// ---------------------------------------------------------------------------
// Fused MoE (gathered) for MI455X / gfx1250, wave32 + bf16 WMMA.
//   T=8192 tokens, topK=2, E=8 experts, D=1024, F=2048, relu^2 activation.
// Pipeline (all on `stream`):
//   1) zero d_out
//   2) transpose+convert w_up / w_down to bf16 (N-major) into d_ws
//   3) deterministic routing compaction -> per-expert slot lists + 64-row tiles
//   4) gather+convert routed token rows to bf16 (slot order) -> Xg
//   5) per-tile fused kernel: H = relu2(Xg @ Wup) staged in 257KB LDS,
//      out += gate * (H @ Wdown) via f32 global atomics.
// ---------------------------------------------------------------------------

#define T_TOK   8192
#define TOPK    2
#define NEXP    8
#define DIM     1024              // hidden dim
#define FDIM    2048              // intermediate dim
#define NSLOT   (T_TOK * TOPK)    // 16384 routed (token,expert) slots
#define BM      64                // rows (tokens) per tile / workgroup
#define H_STRIDE (FDIM + 8)       // LDS row pad: 4112B row stride rotates banks
#define MAX_TILES 264             // sum ceil(cnt_e/64) <= NSLOT/64 + NEXP

typedef __attribute__((ext_vector_type(16))) __bf16 v16bf;
typedef __attribute__((ext_vector_type(8)))  float  v8f;

union BF16x16 { uint4 u[2]; v16bf v; };

__device__ __forceinline__ unsigned short f2bf_bits(float f) {
  union { __bf16 h; unsigned short u; } cv;
  cv.h = (__bf16)f;                 // RNE f32 -> bf16
  return cv.u;
}

// ---------------------------------------------------------------------------
// Kernel 1: zero the f32 output (vectorized b128 stores).
// ---------------------------------------------------------------------------
__global__ void zero_f32_kernel(float4* __restrict__ p, int n4) {
  int i = blockIdx.x * blockDim.x + threadIdx.x;
  if (i < n4) p[i] = make_float4(0.f, 0.f, 0.f, 0.f);
}

// ---------------------------------------------------------------------------
// Kernel 2: tiled transpose f32 [rows][cols] -> bf16 [cols][rows], per expert.
//   grid (cols/32, rows/32, NEXP), block (32, 8)
// ---------------------------------------------------------------------------
__global__ void transpose_f32_to_bf16_kernel(const float* __restrict__ src,
                                             unsigned short* __restrict__ dst,
                                             int rows, int cols) {
  __shared__ float tile[32][33];
  const size_t mat = (size_t)rows * (size_t)cols;
  const float* s = src + (size_t)blockIdx.z * mat;
  unsigned short* d = dst + (size_t)blockIdx.z * mat;
  const int c0 = blockIdx.x * 32, r0 = blockIdx.y * 32;
  const int tx = threadIdx.x, ty = threadIdx.y;
#pragma unroll
  for (int j = 0; j < 32; j += 8)
    tile[ty + j][tx] = s[(size_t)(r0 + ty + j) * cols + (c0 + tx)];
  __syncthreads();
#pragma unroll
  for (int j = 0; j < 32; j += 8)
    d[(size_t)(c0 + ty + j) * rows + (r0 + tx)] = f2bf_bits(tile[tx][ty + j]);
}

// ---------------------------------------------------------------------------
// Kernel 3: deterministic routing compaction (single 1024-thread block).
//   Builds slotList (slot ids grouped by expert, ascending) and 64-row tiles.
// ---------------------------------------------------------------------------
__global__ void route_kernel(const long long* __restrict__ topIdx,
                             int* __restrict__ slotList,
                             int4* __restrict__ tileInfo,
                             int* __restrict__ numTiles) {
  __shared__ int sc[1024];
  __shared__ int sBase;
  __shared__ int expOff[NEXP + 1];
  const int tid = threadIdx.x;
  if (tid == 0) sBase = 0;
  __syncthreads();

  for (int e = 0; e < NEXP; ++e) {
    if (tid == 0) expOff[e] = sBase;
    __syncthreads();
    for (int chunk = 0; chunk < NSLOT; chunk += 1024) {
      const int i = chunk + tid;
      const int flag = (topIdx[i] == (long long)e) ? 1 : 0;
      sc[tid] = flag;
      __syncthreads();
      // Hillis-Steele inclusive scan over 1024
      for (int off = 1; off < 1024; off <<= 1) {
        int v = (tid >= off) ? sc[tid - off] : 0;
        __syncthreads();
        sc[tid] += v;
        __syncthreads();
      }
      const int excl = sc[tid] - flag;
      const int base = sBase;
      if (flag) slotList[base + excl] = i;
      __syncthreads();
      if (tid == 0) sBase = base + sc[1023];
      __syncthreads();
    }
  }
  if (tid == 0) {
    expOff[NEXP] = sBase;   // == NSLOT
    int nt = 0;
    for (int e = 0; e < NEXP; ++e) {
      for (int s = expOff[e]; s < expOff[e + 1]; s += BM) {
        int cnt = expOff[e + 1] - s;
        if (cnt > BM) cnt = BM;
        tileInfo[nt++] = make_int4(s, cnt, e, 0);
      }
    }
    *numTiles = nt;
  }
}

// ---------------------------------------------------------------------------
// Kernel 4: gather routed token rows (slot order) and convert to bf16.
//   grid NSLOT blocks x 256 threads; each thread: 4 consecutive columns.
// ---------------------------------------------------------------------------
__global__ void gather_x_bf16_kernel(const float* __restrict__ X,
                                     const int* __restrict__ slotList,
                                     unsigned short* __restrict__ Xg) {
  const int row = blockIdx.x;                  // 0..NSLOT-1
  const int tok = slotList[row] >> 1;          // slot = token*TOPK + k
  const float* src = X + (size_t)tok * DIM;
  unsigned short* dst = Xg + (size_t)row * DIM;
  const int c = threadIdx.x * 4;
  const float4 f = *(const float4*)(src + c);
  union { unsigned short s[4]; uint2 u; } pk;
  pk.s[0] = f2bf_bits(f.x); pk.s[1] = f2bf_bits(f.y);
  pk.s[2] = f2bf_bits(f.z); pk.s[3] = f2bf_bits(f.w);
  *(uint2*)(dst + c) = pk.u;                   // b64 store
}

// ---------------------------------------------------------------------------
// Kernel 5: fused expert GEMMs. One workgroup (256 thr = 8 wave32) per tile.
//   Phase 1: H[64,2048] = relu2(Xg[64,1024] @ Wup) -> LDS (bf16, padded rows)
//   Phase 2: out += gate * (H @ Wdown), f32 atomics.
// WMMA operand layouts used here (16-bit, contiguous k for A and B):
//   A: lane l holds row (l&15); K = (l>>4)*16 .. +15 contiguous in elems 0..15.
//   B: lane l holds col (l&15); K = (l>>4)*16 .. +15 contiguous in elems 0..15.
//   C: lane l, elem i -> (m = 8*(l>>4)+i, n = l&15).
// ---------------------------------------------------------------------------
__global__ __launch_bounds__(256, 1)
void moe_wmma_kernel(const unsigned short* __restrict__ Xg,   // [NSLOT][D] bf16
                     const float* __restrict__ topW,
                     const unsigned short* __restrict__ wupT,   // [E][F][D] bf16
                     const unsigned short* __restrict__ wdnT,   // [E][D][F] bf16
                     const int* __restrict__ slotList,
                     const int4* __restrict__ tileInfo,
                     const int* __restrict__ numTiles,
                     float* __restrict__ out) {
  extern __shared__ char smem[];
  unsigned short* Hlds = (unsigned short*)smem;                     // BM * H_STRIDE
  int*   tokLds = (int*)(smem + (size_t)BM * H_STRIDE * 2);
  float* wLds   = (float*)(smem + (size_t)BM * H_STRIDE * 2 + BM * 4);

  if ((int)blockIdx.x >= *numTiles) return;
  const int4 ti = tileInfo[blockIdx.x];
  const int rowStart = ti.x, rowCount = ti.y, e = ti.z;

  const int tid = threadIdx.x;
  if (tid < BM) {
    const int slot = (tid < rowCount) ? slotList[rowStart + tid] : -1;
    tokLds[tid] = (slot >= 0) ? (slot >> 1) : 0;
    wLds[tid]   = (slot >= 0) ? topW[slot] : 0.0f;
  }
  __syncthreads();

  const int lane = tid & 31;
  const int wave = tid >> 5;
  const int half = lane >> 4;      // 0 or 1
  const int l15  = lane & 15;
  const int aoff = half * 16;      // A K sub-offset within 32-chunk
  const int boff = half * 16;      // B K sub-offset within 32-chunk

  const unsigned short* Wup = wupT + (size_t)e * FDIM * DIM;
  const unsigned short* Wdn = wdnT + (size_t)e * DIM * FDIM;

  // Per-lane A row pointers (pre-gathered bf16, contiguous in tile).
  const unsigned short* Arow[4];
#pragma unroll
  for (int mt = 0; mt < 4; ++mt) {
    int m = mt * 16 + l15;
    int r = (m < rowCount) ? (rowStart + m) : rowStart;   // clamp padding rows
    Arow[mt] = Xg + (size_t)r * DIM;
  }

  const v8f zacc = {};

  // ---------------- Phase 1: H = relu2(Xg @ Wup), wave owns N-strip of 256 --
  for (int pass = 0; pass < 8; ++pass) {
    const int n0 = wave * 256 + pass * 32;          // two 16-wide N tiles
    v8f acc[2][4];
#pragma unroll
    for (int j = 0; j < 2; ++j)
#pragma unroll
      for (int mt = 0; mt < 4; ++mt) acc[j][mt] = zacc;

    for (int kk = 0; kk < DIM; kk += 32) {
      BF16x16 b[2];
#pragma unroll
      for (int j = 0; j < 2; ++j) {
        const unsigned short* bp =
            Wup + (size_t)(n0 + j * 16 + l15) * DIM + kk + boff;
        b[j].u[0] = *(const uint4*)(bp);
        b[j].u[1] = *(const uint4*)(bp + 8);
      }
      BF16x16 a[4];
#pragma unroll
      for (int mt = 0; mt < 4; ++mt) {
        const unsigned short* ap = Arow[mt] + kk;
        a[mt].u[0] = *(const uint4*)(ap + aoff);
        a[mt].u[1] = *(const uint4*)(ap + aoff + 8);
      }
#pragma unroll
      for (int j = 0; j < 2; ++j)
#pragma unroll
        for (int mt = 0; mt < 4; ++mt)
          acc[j][mt] = __builtin_amdgcn_wmma_f32_16x16x32_bf16(
              false, a[mt].v, false, b[j].v, (short)0, acc[j][mt], false, false);
    }
    // relu^2, write bf16 H tile to LDS
#pragma unroll
    for (int j = 0; j < 2; ++j)
#pragma unroll
      for (int mt = 0; mt < 4; ++mt)
#pragma unroll
        for (int i = 0; i < 8; ++i) {
          float v = acc[j][mt][i];
          v = (v > 0.0f) ? v * v : 0.0f;
          const int m = mt * 16 + half * 8 + i;
          const int n = n0 + j * 16 + l15;
          Hlds[(size_t)m * H_STRIDE + n] = f2bf_bits(v);
        }
  }
  __syncthreads();

  // ---------------- Phase 2: out += gate * (H @ Wdn), wave owns N-strip 128 -
  for (int pass = 0; pass < 4; ++pass) {
    const int n0 = wave * 128 + pass * 32;
    v8f acc[2][4];
#pragma unroll
    for (int j = 0; j < 2; ++j)
#pragma unroll
      for (int mt = 0; mt < 4; ++mt) acc[j][mt] = zacc;

    for (int kk = 0; kk < FDIM; kk += 32) {
      BF16x16 b[2];
#pragma unroll
      for (int j = 0; j < 2; ++j) {
        const unsigned short* bp =
            Wdn + (size_t)(n0 + j * 16 + l15) * FDIM + kk + boff;
        b[j].u[0] = *(const uint4*)(bp);
        b[j].u[1] = *(const uint4*)(bp + 8);
      }
      BF16x16 a[4];
#pragma unroll
      for (int mt = 0; mt < 4; ++mt) {
        const unsigned short* hp =
            Hlds + (size_t)(mt * 16 + l15) * H_STRIDE + kk;
        a[mt].u[0] = *(const uint4*)(hp + aoff);        // ds_load_b128
        a[mt].u[1] = *(const uint4*)(hp + aoff + 8);
      }
#pragma unroll
      for (int j = 0; j < 2; ++j)
#pragma unroll
        for (int mt = 0; mt < 4; ++mt)
          acc[j][mt] = __builtin_amdgcn_wmma_f32_16x16x32_bf16(
              false, a[mt].v, false, b[j].v, (short)0, acc[j][mt], false, false);
    }
    // gate-scale + scatter-add
#pragma unroll
    for (int j = 0; j < 2; ++j)
#pragma unroll
      for (int mt = 0; mt < 4; ++mt)
#pragma unroll
        for (int i = 0; i < 8; ++i) {
          const int m = mt * 16 + half * 8 + i;
          if (m < rowCount) {
            const int n = n0 + j * 16 + l15;
            atomicAdd(&out[(size_t)tokLds[m] * DIM + n],
                      wLds[m] * acc[j][mt][i]);
          }
        }
  }
}

// ---------------------------------------------------------------------------
// Host launcher
// ---------------------------------------------------------------------------
extern "C" void kernel_launch(void* const* d_in, const int* in_sizes, int n_in,
                              void* d_out, int out_size, void* d_ws, size_t ws_size,
                              hipStream_t stream) {
  (void)in_sizes; (void)n_in; (void)out_size; (void)ws_size;

  const float*     X    = (const float*)d_in[0];       // [T, D]
  const long long* topI = (const long long*)d_in[1];   // [T, K] int64
  const float*     topW = (const float*)d_in[2];       // [T, K]
  const float*     wUp  = (const float*)d_in[3];       // [E, D, F]
  const float*     wDn  = (const float*)d_in[4];       // [E, F, D]
  float*           out  = (float*)d_out;               // [T, D]

  // Workspace layout (bytes)
  char* ws = (char*)d_ws;
  const size_t wBytes  = (size_t)NEXP * DIM * FDIM * 2;   // 32 MiB each
  const size_t xgBytes = (size_t)NSLOT * DIM * 2;         // 32 MiB
  unsigned short* wupT = (unsigned short*)(ws);                   // [E][F][D]
  unsigned short* wdnT = (unsigned short*)(ws + wBytes);          // [E][D][F]
  unsigned short* Xg   = (unsigned short*)(ws + 2 * wBytes);      // [NSLOT][D]
  int*  slotList = (int*)(ws + 2 * wBytes + xgBytes);             // 16384 ints
  int4* tileInfo = (int4*)(ws + 2 * wBytes + xgBytes + NSLOT * 4);
  int*  numTiles = (int*)(ws + 2 * wBytes + xgBytes + NSLOT * 4 + MAX_TILES * 16);

  // 1) zero output
  {
    const int n4 = (T_TOK * DIM) / 4;
    zero_f32_kernel<<<(n4 + 255) / 256, 256, 0, stream>>>((float4*)out, n4);
  }
  // 2) weight transpose+convert (per expert via grid.z)
  {
    dim3 blk(32, 8, 1);
    dim3 gUp(FDIM / 32, DIM / 32, NEXP);   // w_up:   [D][F] -> [F][D]
    transpose_f32_to_bf16_kernel<<<gUp, blk, 0, stream>>>(wUp, wupT, DIM, FDIM);
    dim3 gDn(DIM / 32, FDIM / 32, NEXP);   // w_down: [F][D] -> [D][F]
    transpose_f32_to_bf16_kernel<<<gDn, blk, 0, stream>>>(wDn, wdnT, FDIM, DIM);
  }
  // 3) routing compaction (deterministic, single block)
  route_kernel<<<1, 1024, 0, stream>>>(topI, slotList, tileInfo, numTiles);

  // 4) gather routed rows to bf16 in slot order
  gather_x_bf16_kernel<<<NSLOT, 256, 0, stream>>>(X, slotList, Xg);

  // 5) fused expert GEMMs
  {
    const int ldsBytes = BM * H_STRIDE * 2 + BM * 4 + BM * 4;  // 263,680 B
    moe_wmma_kernel<<<MAX_TILES, 256, ldsBytes, stream>>>(
        Xg, topW, wupT, wdnT, slotList, tileInfo, numTiles, out);
  }
}
